// LGBlock_26912265076825
// MI455X (gfx1250) — matrix operand order repaired and checked by reading, not compiled, not run on hardware
//
#include <hip/hip_runtime.h>
#include <hip/hip_bf16.h>

// ---------------------------------------------------------------------------
// Transformer block for MI455X (gfx1250, wave32).
// - All GEMM-like math on v_wmma_f32_16x16x32_f16 (f16 in, f32 accumulate).
// - All global->LDS tile staging via global_load_async_to_lds_b128 (ASYNCcnt),
//   double-buffered so DMA of tile k+1 overlaps WMMA on tile k.
// - Weights pre-transposed to f16 so every DMA is contiguous and every
//   LDS fragment read is a pair of ds_load_b128.
// ---------------------------------------------------------------------------

typedef __attribute__((ext_vector_type(16))) _Float16 v16h;
typedef __attribute__((ext_vector_type(8)))  float    v8f;

constexpr int D  = 512;
constexpr int H  = 8;
constexpr int DH = 64;
constexpr int WIN = 128;
constexpr int FF = 2048;
constexpr int B  = 4;
constexpr int T  = 2048;
constexpr float EPS = 1e-5f;
constexpr float NEG_BIG = -3.0e38f;

static __device__ __forceinline__ v8f wmma32(v16h a, v16h b, v8f c) {
  return __builtin_amdgcn_wmma_f32_16x16x32_f16(false, a, false, b,
                                                (short)0, c, false, false);
}

static __device__ __forceinline__ v8f zero8() {
  v8f z;
#pragma unroll
  for (int i = 0; i < 8; ++i) z[i] = 0.0f;
  return z;
}

// A-fragment (16x32 f16) element j -> K index: j + 16*half (one contiguous
// 16-element run per lane, same order as B).
static __device__ __forceinline__ int a_kk(int j, int half) {
  return j + (half << 4);
}
// B-fragment element j -> K index: j + 16*half (fully contiguous run).

// CDNA5 async DMA: memory -> LDS, 16B per lane, tracked by ASYNCcnt.
static __device__ __forceinline__ void async_b128(unsigned lds_byte,
                                                  unsigned gbyte,
                                                  const void* sbase) {
  asm volatile("global_load_async_to_lds_b128 %0, %1, %2"
               :: "v"(lds_byte), "v"(gbyte), "s"(sbase) : "memory");
}
static __device__ __forceinline__ void wait_async0() {
  asm volatile("s_wait_asynccnt 0x0" ::: "memory");
}
static __device__ __forceinline__ unsigned lds_addr(const void* p) {
  return (unsigned)(size_t)p;   // LDS aperture: addr[31:0] is the LDS offset
}

// ---------------------------------------------------------------------------
// fp32 -> f16 transposed weight conversion: Wt[n*K + k] = W[k*N + n]
// ---------------------------------------------------------------------------
__global__ __launch_bounds__(256) void k_cvt_t(const float* __restrict__ s,
                                               _Float16* __restrict__ d,
                                               int rowsK, int colsN) {
  int i = blockIdx.x * 256 + threadIdx.x;
  if (i < rowsK * colsN) {
    int r = i / colsN, c = i % colsN;
    d[(size_t)c * rowsK + r] = (_Float16)s[i];
  }
}

// ---------------------------------------------------------------------------
// LayerNorm + RoPE fused; one block per (b,t) row; thread owns (tid, tid+256)
// which is exactly the RoPE pair.
// ---------------------------------------------------------------------------
__global__ __launch_bounds__(256) void k_ln_rope(const float* __restrict__ x,
                                                 const float* __restrict__ g,
                                                 const float* __restrict__ bta,
                                                 _Float16* __restrict__ h) {
  __shared__ float red[16];
  __shared__ float st[2];
  int row = blockIdx.x, tid = threadIdx.x;
  const float* xr = x + (size_t)row * D;
  float a = xr[tid], b = xr[tid + 256];
  float s = a + b, q = a * a + b * b;
#pragma unroll
  for (int d2 = 16; d2 >= 1; d2 >>= 1) {
    s += __shfl_xor(s, d2, 32);
    q += __shfl_xor(q, d2, 32);
  }
  if ((tid & 31) == 0) { red[tid >> 5] = s; red[8 + (tid >> 5)] = q; }
  __syncthreads();
  if (tid == 0) {
    float S = 0.f, Q = 0.f;
#pragma unroll
    for (int i = 0; i < 8; ++i) { S += red[i]; Q += red[8 + i]; }
    float mu = S * (1.0f / D);
    float var = Q * (1.0f / D) - mu * mu;
    st[0] = mu; st[1] = rsqrtf(var + EPS);
  }
  __syncthreads();
  float mu = st[0], rs = st[1];
  float y1 = (a - mu) * rs * g[tid] + bta[tid];
  float y2 = (b - mu) * rs * g[tid + 256] + bta[tid + 256];
  int t = row % T;
  float inv = __expf(-(float)tid * (9.210340371976184f / 256.0f));
  float ang = (float)t * inv;
  float sn, cs;
  __sincosf(ang, &sn, &cs);
  h[(size_t)row * D + tid]       = (_Float16)(y1 * cs - y2 * sn);
  h[(size_t)row * D + tid + 256] = (_Float16)(y1 * sn + y2 * cs);
}

// ---------------------------------------------------------------------------
// Double LayerNorm (pre-FFN): z = LN2(LN1(x)), f16 out.
// ---------------------------------------------------------------------------
__global__ __launch_bounds__(256) void k_ln_ln(const float* __restrict__ x,
                                               const float* __restrict__ g1,
                                               const float* __restrict__ bb1,
                                               const float* __restrict__ g2,
                                               const float* __restrict__ bb2,
                                               _Float16* __restrict__ o) {
  __shared__ float red[16];
  __shared__ float st[2];
  int row = blockIdx.x, tid = threadIdx.x;
  const float* xr = x + (size_t)row * D;
  float a = xr[tid], b = xr[tid + 256];

  float s = a + b, q = a * a + b * b;
#pragma unroll
  for (int d2 = 16; d2 >= 1; d2 >>= 1) { s += __shfl_xor(s, d2, 32); q += __shfl_xor(q, d2, 32); }
  if ((tid & 31) == 0) { red[tid >> 5] = s; red[8 + (tid >> 5)] = q; }
  __syncthreads();
  if (tid == 0) {
    float S = 0.f, Q = 0.f;
#pragma unroll
    for (int i = 0; i < 8; ++i) { S += red[i]; Q += red[8 + i]; }
    float mu = S * (1.0f / D), var = Q * (1.0f / D) - mu * mu;
    st[0] = mu; st[1] = rsqrtf(var + EPS);
  }
  __syncthreads();
  float y1 = (a - st[0]) * st[1] * g1[tid] + bb1[tid];
  float y2 = (b - st[0]) * st[1] * g1[tid + 256] + bb1[tid + 256];
  __syncthreads();

  s = y1 + y2; q = y1 * y1 + y2 * y2;
#pragma unroll
  for (int d2 = 16; d2 >= 1; d2 >>= 1) { s += __shfl_xor(s, d2, 32); q += __shfl_xor(q, d2, 32); }
  if ((tid & 31) == 0) { red[tid >> 5] = s; red[8 + (tid >> 5)] = q; }
  __syncthreads();
  if (tid == 0) {
    float S = 0.f, Q = 0.f;
#pragma unroll
    for (int i = 0; i < 8; ++i) { S += red[i]; Q += red[8 + i]; }
    float mu = S * (1.0f / D), var = Q * (1.0f / D) - mu * mu;
    st[0] = mu; st[1] = rsqrtf(var + EPS);
  }
  __syncthreads();
  float z1 = (y1 - st[0]) * st[1] * g2[tid] + bb2[tid];
  float z2 = (y2 - st[0]) * st[1] * g2[tid + 256] + bb2[tid + 256];
  o[(size_t)row * D + tid]       = (_Float16)z1;
  o[(size_t)row * D + tid + 256] = (_Float16)z2;
}

// ---------------------------------------------------------------------------
// Tiled WMMA GEMM: C[M,N] = A[M,K] * Wt[N,K]^T + bias.
// Block 256 thr (8 waves), 128x128 tile; wave tile 32x64 (8 accumulators).
// Double-buffered async DMA staging. LDS rows padded to 40 halfs so the 16
// lanes of a fragment read hit 16 distinct banks.
// ---------------------------------------------------------------------------
enum { EPI_QKV3 = 0, EPI_RES = 1, EPI_GELU = 2 };

template <int EPI>
__global__ __launch_bounds__(256) void k_gemm(const _Float16* __restrict__ A,
                                              const _Float16* __restrict__ Wt,
                                              const float* __restrict__ bias0,
                                              const float* __restrict__ bias1,
                                              const float* __restrict__ bias2,
                                              const float* __restrict__ resid,
                                              float* __restrict__ outf,
                                              _Float16* __restrict__ outh,
                                              int M, int N, int K) {
  __shared__ __align__(16) _Float16 lsA[2][128 * 40];
  __shared__ __align__(16) _Float16 lsB[2][128 * 40];
  int tid = threadIdx.x;
  int wave = tid >> 5, lane = tid & 31;
  int wr = wave & 3, wc = wave >> 2;           // 4x2 grid of 32x64 wave tiles
  int row0 = blockIdx.y * 128, col0 = blockIdx.x * 128;
  int m = lane & 15, half = lane >> 4;

  // Each thread DMAs 2 b128 of A and 2 b128 of B per buffer (128x32 each).
  auto issue = [&](int buf, int k0) {
#pragma unroll
    for (int u = 0; u < 2; ++u) {
      int t = tid + u * 256;
      int r = t >> 2, sg = t & 3;
      async_b128(lds_addr(&lsA[buf][r * 40 + sg * 8]),
                 (unsigned)(((size_t)(row0 + r) * K + k0 + sg * 8) * 2), A);
      async_b128(lds_addr(&lsB[buf][r * 40 + sg * 8]),
                 (unsigned)(((size_t)(col0 + r) * K + k0 + sg * 8) * 2), Wt);
    }
  };

  v8f acc[2][4];
#pragma unroll
  for (int s2 = 0; s2 < 2; ++s2)
#pragma unroll
    for (int c = 0; c < 4; ++c) acc[s2][c] = zero8();

  int nk = K / 32;
  issue(0, 0);
  for (int kb = 0; kb < nk; ++kb) {
    int cur = kb & 1;
    wait_async0();        // own batch for buffer `cur` is the only one inflight
    __syncthreads();      // everyone's `cur` is in LDS; prior readers of `nxt` done
    if (kb + 1 < nk) issue(cur ^ 1, (kb + 1) * 32);   // DMA overlaps compute

    const _Float16* la = &lsA[cur][0];
    const _Float16* lb = &lsB[cur][0];
    v16h af[2];
#pragma unroll
    for (int sub = 0; sub < 2; ++sub)
#pragma unroll
      for (int j = 0; j < 16; ++j)
        af[sub][j] = la[(wr * 32 + sub * 16 + m) * 40 + a_kk(j, half)];
#pragma unroll
    for (int c = 0; c < 4; ++c) {
      v16h bf;
#pragma unroll
      for (int j = 0; j < 16; ++j)
        bf[j] = lb[(wc * 64 + c * 16 + m) * 40 + half * 16 + j];
      acc[0][c] = wmma32(af[0], bf, acc[0][c]);
      acc[1][c] = wmma32(af[1], bf, acc[1][c]);
    }
  }

#pragma unroll
  for (int sub = 0; sub < 2; ++sub)
#pragma unroll
    for (int r = 0; r < 8; ++r) {
      int gr = row0 + wr * 32 + sub * 16 + r + half * 8;
#pragma unroll
      for (int c2 = 0; c2 < 4; ++c2) {
        int gc = col0 + wc * 64 + c2 * 16 + m;
        float cv = acc[sub][c2][r];
        if (EPI == EPI_QKV3) {
          // N = 3*D: which 0=Q,1=K ([B,H,T,DH]) ; 2=V transposed ([B,H,DH,T])
          int which = gc >> 9, col = gc & 511;
          int hh = col >> 6, dh = col & (DH - 1);
          const float* bias = which == 0 ? bias0 : (which == 1 ? bias1 : bias2);
          cv += bias[col];
          int bb = gr / T, t = gr % T;
          size_t dst;
          if (which < 2)
            dst = (size_t)which * ((size_t)B * T * D) +
                  (((size_t)bb * H + hh) * T + t) * DH + dh;
          else
            dst = 2 * ((size_t)B * T * D) +
                  (((size_t)bb * H + hh) * DH + dh) * (size_t)T + t;
          outh[dst] = (_Float16)cv;
        } else if (EPI == EPI_RES) {
          cv += bias0[gc];
          outf[(size_t)gr * N + gc] = resid[(size_t)gr * N + gc] + cv;
        } else {  // EPI_GELU exact
          cv += bias0[gc];
          float gl = 0.5f * cv * (1.0f + erff(cv * 0.70710678118654752f));
          outh[(size_t)gr * N + gc] = (_Float16)gl;
        }
      }
    }
}

// ---------------------------------------------------------------------------
// Attention, anti-window mask (|q-k| > WIN allowed), flash-style online
// softmax. Wave owns a 16-query tile; 8 waves share double-buffered async
// K / V^T tiles (32 keys per step). V supplied transposed [B,H,DH,T].
// ---------------------------------------------------------------------------
__global__ __launch_bounds__(256) void k_attn(const _Float16* __restrict__ q,
                                              const _Float16* __restrict__ k,
                                              const _Float16* __restrict__ vt,
                                              _Float16* __restrict__ o) {
  __shared__ __align__(16) _Float16 lsK[2][32 * 72];   // [key][dh], padded
  __shared__ __align__(16) _Float16 lsVt[2][64 * 40];  // [dh][key], padded
  __shared__ __align__(16) _Float16 lsP[8][16 * 40];   // per-wave P tile

  int tid = threadIdx.x, wave = tid >> 5, lane = tid & 31;
  int m = lane & 15, half = lane >> 4;
  int bh = blockIdx.y;
  int b = bh >> 3, hh = bh & 7;
  int q0 = blockIdx.x * 128 + wave * 16;
  size_t kbase = (size_t)bh * T * DH;    // k:  [bh][t][dh]
  size_t vbase = (size_t)bh * DH * T;    // vt: [bh][dh][t]

  auto issueKV = [&](int buf, int j0) {
    int r = tid >> 3, sg = tid & 7;      // K tile: 32 rows x 64 dh
    async_b128(lds_addr(&lsK[buf][r * 72 + sg * 8]),
               (unsigned)((kbase + (size_t)(j0 + r) * DH + sg * 8) * 2), k);
    int r2 = tid >> 2, s2 = tid & 3;     // Vt tile: 64 dh-rows x 32 keys
    async_b128(lds_addr(&lsVt[buf][r2 * 40 + s2 * 8]),
               (unsigned)((vbase + (size_t)r2 * T + j0 + s2 * 8) * 2), vt);
  };

  // Q fragments (16x64 => two 16x32 A-frags), loaded once from global.
  v16h qa0, qa1;
#pragma unroll
  for (int j = 0; j < 16; ++j) {
    int kk = a_kk(j, half);
    qa0[j] = q[kbase + (size_t)(q0 + m) * DH + kk];
    qa1[j] = q[kbase + (size_t)(q0 + m) * DH + 32 + kk];
  }

  v8f oa[4];
#pragma unroll
  for (int c = 0; c < 4; ++c) oa[c] = zero8();
  float mr[8], lr[8];
#pragma unroll
  for (int r = 0; r < 8; ++r) { mr[r] = NEG_BIG; lr[r] = 0.0f; }

  issueKV(0, 0);
  for (int j0 = 0; j0 < T; j0 += 32) {
    int cur = (j0 >> 5) & 1;
    wait_async0();
    __syncthreads();
    if (j0 + 32 < T) issueKV(cur ^ 1, j0 + 32);

    // S = Q K^T : two 16-key accumulators, K-dim 64 in two chunks
    v8f sAcc[2];
#pragma unroll
    for (int sub = 0; sub < 2; ++sub) {
      v8f s = zero8();
#pragma unroll
      for (int ch = 0; ch < 2; ++ch) {
        v16h kbf;
#pragma unroll
        for (int j = 0; j < 16; ++j)
          kbf[j] = lsK[cur][(sub * 16 + m) * 72 + ch * 32 + half * 16 + j];
        s = wmma32(ch ? qa1 : qa0, kbf, s);
      }
      sAcc[sub] = s;
    }

    // Online softmax; row (r+8*half) lives in one 16-lane group.
#pragma unroll
    for (int r = 0; r < 8; ++r) {
      int qi = q0 + r + half * 8;
      int k0i = j0 + m, k1i = j0 + 16 + m;
      int d0 = qi - k0i; if (d0 < 0) d0 = -d0;
      int d1 = qi - k1i; if (d1 < 0) d1 = -d1;
      bool a0 = d0 > WIN, a1 = d1 > WIN;
      float s0v = sAcc[0][r] * 0.125f;
      float s1v = sAcc[1][r] * 0.125f;
      float mx = fmaxf(a0 ? s0v : NEG_BIG, a1 ? s1v : NEG_BIG);
#pragma unroll
      for (int d2 = 8; d2 >= 1; d2 >>= 1) mx = fmaxf(mx, __shfl_xor(mx, d2, 32));
      float mn = fmaxf(mr[r], mx);
      float sc = (mn == mr[r]) ? 1.0f : __expf(mr[r] - mn);
      float p0 = a0 ? __expf(s0v - mn) : 0.0f;
      float p1 = a1 ? __expf(s1v - mn) : 0.0f;
      float ps = p0 + p1;
#pragma unroll
      for (int d2 = 8; d2 >= 1; d2 >>= 1) ps += __shfl_xor(ps, d2, 32);
      lr[r] = lr[r] * sc + ps;
      mr[r] = mn;
#pragma unroll
      for (int c = 0; c < 4; ++c) oa[c][r] *= sc;
      lsP[wave][(r + half * 8) * 40 + m]      = (_Float16)p0;
      lsP[wave][(r + half * 8) * 40 + 16 + m] = (_Float16)p1;
    }

    // Same-wave DS store->load relayout of P: DS in-order; make explicit.
    asm volatile("s_wait_dscnt 0x0" ::: "memory");

    // O += P(16x32) * V(32x64)
    v16h pa;
#pragma unroll
    for (int j = 0; j < 16; ++j) pa[j] = lsP[wave][m * 40 + a_kk(j, half)];
#pragma unroll
    for (int c = 0; c < 4; ++c) {
      v16h vb;
#pragma unroll
      for (int j = 0; j < 16; ++j)
        vb[j] = lsVt[cur][(c * 16 + m) * 40 + half * 16 + j];
      oa[c] = wmma32(pa, vb, oa[c]);
    }
  }

  // Normalize and recombine heads -> o16 [B,T,D]
#pragma unroll
  for (int r = 0; r < 8; ++r) {
    int t = q0 + r + half * 8;
    float inv = 1.0f / lr[r];
#pragma unroll
    for (int c = 0; c < 4; ++c)
      o[((size_t)b * T + t) * D + hh * DH + c * 16 + m] = (_Float16)(oa[c][r] * inv);
  }
}

// ---------------------------------------------------------------------------
// Launcher
// ---------------------------------------------------------------------------
extern "C" void kernel_launch(void* const* d_in, const int* in_sizes, int n_in,
                              void* d_out, int out_size, void* d_ws, size_t ws_size,
                              hipStream_t stream) {
  (void)in_sizes; (void)n_in; (void)out_size; (void)ws_size;
  const float* x    = (const float*)d_in[0];
  const float* Wq   = (const float*)d_in[1];
  const float* bq   = (const float*)d_in[2];
  const float* Wk   = (const float*)d_in[3];
  const float* bk   = (const float*)d_in[4];
  const float* Wv   = (const float*)d_in[5];
  const float* bv   = (const float*)d_in[6];
  const float* Wo   = (const float*)d_in[7];
  const float* bo   = (const float*)d_in[8];
  const float* ln_g = (const float*)d_in[9];
  const float* ln_b = (const float*)d_in[10];
  const float* ffg  = (const float*)d_in[11];
  const float* ffb  = (const float*)d_in[12];
  const float* W1   = (const float*)d_in[13];
  const float* b1   = (const float*)d_in[14];
  const float* W2   = (const float*)d_in[15];
  const float* b2   = (const float*)d_in[16];
  float* out = (float*)d_out;

  char* ws = (char*)d_ws;
  size_t off = 0;
  auto take = [&](size_t bytes) -> char* {
    char* p = ws + off;
    off += (bytes + 255) & ~(size_t)255;
    return p;
  };
  const size_t BTD = (size_t)B * T * D;

  _Float16* h16   = (_Float16*)take(BTD * 2);            // LN+RoPE out [M,D]
  _Float16* qkv16 = (_Float16*)take(3 * BTD * 2);        // Q,K [B,H,T,DH]; V^T [B,H,DH,T]
  _Float16* o16   = (_Float16*)take(BTD * 2);            // attn out [B,T,D]
  float*    x1    = (float*)take(BTD * 4);               // x + attn projection
  _Float16* f16a  = (_Float16*)take(BTD * 2);            // double-LN out
  _Float16* u16   = (_Float16*)take((size_t)B * T * FF * 2);
  _Float16* Wqkvt = (_Float16*)take((size_t)3 * D * D * 2);  // [3D rows][K=D]
  _Float16* Wot   = (_Float16*)take((size_t)D * D * 2);      // [D][D]
  _Float16* W1t   = (_Float16*)take((size_t)FF * D * 2);     // [FF][D]
  _Float16* W2t   = (_Float16*)take((size_t)D * FF * 2);     // [D][FF]

  const int M = B * T;
  const int nDD = D * D, nDF = D * FF;

  // 1) weights -> f16, transposed [N][K]; QKV stacked along N
  k_cvt_t<<<(nDD + 255) / 256, 256, 0, stream>>>(Wq, Wqkvt + 0 * (size_t)nDD, D, D);
  k_cvt_t<<<(nDD + 255) / 256, 256, 0, stream>>>(Wk, Wqkvt + 1 * (size_t)nDD, D, D);
  k_cvt_t<<<(nDD + 255) / 256, 256, 0, stream>>>(Wv, Wqkvt + 2 * (size_t)nDD, D, D);
  k_cvt_t<<<(nDD + 255) / 256, 256, 0, stream>>>(Wo, Wot, D, D);
  k_cvt_t<<<(nDF + 255) / 256, 256, 0, stream>>>(W1, W1t, D, FF);
  k_cvt_t<<<(nDF + 255) / 256, 256, 0, stream>>>(W2, W2t, FF, D);

  // 2) LN + RoPE
  k_ln_rope<<<M, 256, 0, stream>>>(x, ln_g, ln_b, h16);

  // 3) fused QKV projection (one pass over activations, N = 3D)
  dim3 gqkv(3 * D / 128, M / 128);
  k_gemm<EPI_QKV3><<<gqkv, 256, 0, stream>>>(h16, Wqkvt, bq, bk, bv,
                                             nullptr, nullptr, qkv16, M, 3 * D, D);

  // 4) masked attention
  dim3 ga(T / 128, B * H);
  k_attn<<<ga, 256, 0, stream>>>(qkv16, qkv16 + BTD, qkv16 + 2 * BTD, o16);

  // 5) output projection + residual (f32)
  dim3 go(D / 128, M / 128);
  k_gemm<EPI_RES><<<go, 256, 0, stream>>>(o16, Wot, bo, nullptr, nullptr,
                                          x, x1, nullptr, M, D, D);

  // 6) double LN
  k_ln_ln<<<M, 256, 0, stream>>>(x1, ln_g, ln_b, ffg, ffb, f16a);

  // 7) FFN1 + exact GELU
  dim3 g1(FF / 128, M / 128);
  k_gemm<EPI_GELU><<<g1, 256, 0, stream>>>(f16a, W1t, b1, nullptr, nullptr,
                                           nullptr, nullptr, u16, M, FF, D);

  // 8) FFN2 + residual -> f32 output
  dim3 g2(D / 128, M / 128);
  k_gemm<EPI_RES><<<g2, 256, 0, stream>>>(u16, W2t, b2, nullptr, nullptr,
                                          x1, out, nullptr, M, D, FF);
}
